// NaiveFourierKANLayer_463856468096
// MI455X (gfx1250) — matrix operand order repaired and checked by reading, not compiled, not run on hardware
//
#include <hip/hip_runtime.h>
#include <math.h>

typedef __bf16  bf16x16 __attribute__((ext_vector_type(16)));
typedef float   f32x8   __attribute__((ext_vector_type(8)));
typedef float   f32x4   __attribute__((ext_vector_type(4)));

#define N_ROWS   2048
#define I_DIM    256
#define J_DIM    256
#define K_GRID   300
#define R_TOT    (I_DIM * K_GRID)   // 76800  (flat reduction length per d)
#define CHUNKS_TOT (R_TOT / 32)     // 2400
#define BLK_M    128                // rows per workgroup (8 waves x 16)
#define BLK_J    64                 // cols per workgroup
#define WAVE_M   16
#define JT_PER_WAVE 4               // 4 j-tiles of 16 => 64 cols per wave

// Fused cos/sin + bf16 WMMA GEMM over flat reduction r = i*300 + k.
// dst is either the output (direct=1, bias added here) or a per-segment
// partial buffer of size gridDim.z * N_ROWS * J_DIM floats (direct=0).
__global__ void __launch_bounds__(256, 1)
fkan_main(const float* __restrict__ xg, const float* __restrict__ cg,
          const float* __restrict__ biasg, float* __restrict__ dst,
          int chunksPerSeg, int direct)
{
    // [buf][d(cos/sin)][khalf][j][16 bf16]  : 2 * 2*2*64*32B = 16 KB
    __shared__ alignas(32) __bf16 ldsB[2][2][2][BLK_J][16];

    const int tid  = threadIdx.x;
    const int lane = tid & 31;
    const int wave = tid >> 5;
    const int m    = lane & 15;      // row within 16x16 tile / B column
    const int hb   = lane >> 4;      // lane-half: selects K sub-range
    const int kb   = hb * 16;        // A-fragment K base (contiguous: k = 16*hb + s)

    const int rowBase = blockIdx.x * BLK_M + wave * WAVE_M;
    const int jBase   = blockIdx.y * BLK_J;
    const int seg     = blockIdx.z;

    int r0  = seg * chunksPerSeg * 32;     // flat reduction offset of chunk
    int i0  = r0 / K_GRID;                 // uniform scalar state
    int kg0 = r0 - i0 * K_GRID;

    // ---- cooperative B-fill role: one 16-float row per thread ----
    const int f_d  = (tid >> 7) & 1;       // cos/sin coeff plane
    const int f_kh = (tid >> 6) & 1;       // K half (0..15 / 16..31)
    const int f_jj = tid & 63;             // column within block
    const float* fsrc = cg + (size_t)(f_d * J_DIM + (jBase + f_jj)) * (size_t)R_TOT
                           + (size_t)(f_kh * 16);

    f32x8 acc[JT_PER_WAVE];
    #pragma unroll
    for (int jt = 0; jt < JT_PER_WAVE; ++jt)
        #pragma unroll
        for (int e = 0; e < 8; ++e) acc[jt][e] = 0.0f;

    // prefetch chunk 0 (64B contiguous per thread, always in-bounds)
    f32x4 g0, g1, g2, g3;
    {
        const float* p = fsrc + r0;
        g0 = *(const f32x4*)(p + 0);  g1 = *(const f32x4*)(p + 4);
        g2 = *(const f32x4*)(p + 8);  g3 = *(const f32x4*)(p + 12);
    }

    const float* xrow = xg + (size_t)(rowBase + m) * I_DIM;

    int cur = 0;
    for (int cc = 0; cc < chunksPerSeg; ++cc) {
        // ---- convert staged fp32 -> bf16, store to LDS ----
        {
            bf16x16 bv;
            bv[0]=(__bf16)g0.x;  bv[1]=(__bf16)g0.y;  bv[2]=(__bf16)g0.z;  bv[3]=(__bf16)g0.w;
            bv[4]=(__bf16)g1.x;  bv[5]=(__bf16)g1.y;  bv[6]=(__bf16)g1.z;  bv[7]=(__bf16)g1.w;
            bv[8]=(__bf16)g2.x;  bv[9]=(__bf16)g2.y;  bv[10]=(__bf16)g2.z; bv[11]=(__bf16)g2.w;
            bv[12]=(__bf16)g3.x; bv[13]=(__bf16)g3.y; bv[14]=(__bf16)g3.z; bv[15]=(__bf16)g3.w;
            *(bf16x16*)&ldsB[cur][f_d][f_kh][f_jj][0] = bv;
        }
        __syncthreads();

        // ---- issue next chunk's global loads before compute (overlap) ----
        if (cc + 1 < chunksPerSeg) {
            const float* p = fsrc + (r0 + 32);
            g0 = *(const f32x4*)(p + 0);  g1 = *(const f32x4*)(p + 4);
            g2 = *(const f32x4*)(p + 8);  g3 = *(const f32x4*)(p + 12);
        }

        // ---- A fragments: cos/sin of x[b,i]*(k+1), contiguous k = 16*hb + s order ----
        // Trans-pipe is the scarce resource: use 6 transcendentals per chunk
        // (two set bases + step) and exact angle-addition rotations (full-rate
        // FMAs) for the other 14 slot pairs. Rotation error << bf16 rounding.
        float cosv[16], sinv[16];
        const float xv0 = xrow[i0];
        if (kg0 < 269) {               // uniform: no i-boundary in this chunk
            const float cx = __cosf(xv0);
            const float sx = __sinf(xv0);
            float a1 = xv0 * (float)(kg0 + kb + 1);
            float a2 = xv0 * (float)(kg0 + kb + 9);
            float c1 = __cosf(a1), s1 = __sinf(a1);
            float c2 = __cosf(a2), s2 = __sinf(a2);
            cosv[0] = c1; sinv[0] = s1;
            cosv[8] = c2; sinv[8] = s2;
            #pragma unroll
            for (int s = 1; s < 8; ++s) {
                const float nc1 = fmaf(-s1, sx, c1 * cx);
                const float ns1 = fmaf( c1, sx, s1 * cx);
                const float nc2 = fmaf(-s2, sx, c2 * cx);
                const float ns2 = fmaf( c2, sx, s2 * cx);
                c1 = nc1; s1 = ns1; c2 = nc2; s2 = ns2;
                cosv[s]     = c1; sinv[s]     = s1;
                cosv[s + 8] = c2; sinv[s + 8] = s2;
            }
        } else {                        // chunk spans i0 -> i0+1 (~11% of chunks)
            const float xv1 = xrow[i0 + 1];
            const int rbK = K_GRID - kg0;   // K >= rbK belongs to i0+1
            #pragma unroll
            for (int s = 0; s < 16; ++s) {
                const int K = kb + s;
                const bool hibnd = (K >= rbK);
                const float xs = hibnd ? xv1 : xv0;
                const int   f  = K + 1 + (hibnd ? (kg0 - K_GRID) : kg0);
                const float ang = xs * (float)f;
                cosv[s] = __cosf(ang); sinv[s] = __sinf(ang);
            }
        }
        bf16x16 Ac, As;
        #pragma unroll
        for (int s = 0; s < 16; ++s) { Ac[s] = (__bf16)cosv[s]; As[s] = (__bf16)sinv[s]; }

        // ---- 8 WMMAs per chunk: 4 j-tiles x (cos,sin) ----
        #pragma unroll
        for (int jt = 0; jt < JT_PER_WAVE; ++jt) {
            const int jj = jt * 16 + m;
            bf16x16 Bc = *(const bf16x16*)&ldsB[cur][0][hb][jj][0];
            bf16x16 Bs = *(const bf16x16*)&ldsB[cur][1][hb][jj][0];
            acc[jt] = __builtin_amdgcn_wmma_f32_16x16x32_bf16(
                          false, Ac, false, Bc, (short)0, acc[jt], false, false);
            acc[jt] = __builtin_amdgcn_wmma_f32_16x16x32_bf16(
                          false, As, false, Bs, (short)0, acc[jt], false, false);
        }
        __syncthreads();

        cur ^= 1;
        r0  += 32;
        kg0 += 32;
        if (kg0 >= K_GRID) { kg0 -= K_GRID; ++i0; }
    }

    // ---- write C (ISA 32-bit C/D layout: row = v + 8*hb, col = m) ----
    const size_t segOff = (size_t)seg * ((size_t)N_ROWS * J_DIM);
    #pragma unroll
    for (int jt = 0; jt < JT_PER_WAVE; ++jt) {
        const int col = jBase + jt * 16 + m;
        #pragma unroll
        for (int v = 0; v < 8; ++v) {
            const size_t off = (size_t)(rowBase + hb * 8 + v) * J_DIM + col;
            if (direct) dst[off] = acc[jt][v] + biasg[col];
            else        dst[segOff + off] = acc[jt][v];
        }
    }
}

// Deterministic partial-sum reduction + bias (fixed summation order).
__global__ void __launch_bounds__(256)
fkan_reduce(const float* __restrict__ partial, const float* __restrict__ biasg,
            float* __restrict__ out, int S)
{
    const int idx = blockIdx.x * 256 + threadIdx.x;
    const int total = N_ROWS * J_DIM;
    if (idx < total) {
        float s = biasg[idx & (J_DIM - 1)];
        for (int p = 0; p < S; ++p) s += partial[(size_t)p * total + idx];
        out[idx] = s;
    }
}

extern "C" void kernel_launch(void* const* d_in, const int* in_sizes, int n_in,
                              void* d_out, int out_size, void* d_ws, size_t ws_size,
                              hipStream_t stream)
{
    const float* x      = (const float*)d_in[0];
    const float* coeffs = (const float*)d_in[1];
    const float* bias   = (const float*)d_in[2];
    float*       out    = (float*)d_out;

    const size_t outElems = (size_t)N_ROWS * J_DIM;

    // K-split factor chosen by available scratch (needed for occupancy:
    // S=8 -> 512 WGs / 4096 waves; S=4 -> 256 WGs; without split only 64 WGs).
    int S = 1;
    if      (ws_size >= 8 * outElems * sizeof(float)) S = 8;
    else if (ws_size >= 4 * outElems * sizeof(float)) S = 4;
    else if (ws_size >= 2 * outElems * sizeof(float)) S = 2;

    dim3 grid(N_ROWS / BLK_M, J_DIM / BLK_J, S);

    if (S == 1) {
        fkan_main<<<grid, 256, 0, stream>>>(x, coeffs, bias, out, CHUNKS_TOT, 1);
    } else {
        float* partial = (float*)d_ws;
        fkan_main<<<grid, 256, 0, stream>>>(x, coeffs, bias, partial, CHUNKS_TOT / S, 0);
        fkan_reduce<<<(int)((outElems + 255) / 256), 256, 0, stream>>>(partial, bias, out, S);
    }
}
